// Qwen3MoeSparseMoeBlock_76295799046237
// MI455X (gfx1250) — matrix operand order repaired and checked by reading, not compiled, not run on hardware
//
#include <hip/hip_runtime.h>
#include <hip/hip_bf16.h>

// Problem constants (fixed by the reference).
#define TQ 1024
#define DQ 2048
#define EQ 16
#define KQ 4
#define FQ 768

typedef __attribute__((ext_vector_type(16))) __bf16 v16bf;
typedef __attribute__((ext_vector_type(8)))  __bf16 v8bf;
typedef __attribute__((ext_vector_type(8)))  float  v8f;

#define LDAL 72  // padded LDS row stride in ushorts for K=64 stage (144B, 16B-aligned)

// Workspace layout (bytes). Total ~29.5 MB.
#define X_OFF    ((size_t)0)                       // bf16 x: T*D*2 = 4 MB
#define CNT_OFF  ((size_t)(TQ * DQ * 2))           // E ints (padded 256B)
#define LIST_OFF (CNT_OFF + 256)                   // E*T ints = 64 KB
#define RW_OFF   (LIST_OFF + (size_t)EQ * TQ * 4)  // E*T floats = 64 KB
#define H_OFF    (RW_OFF + (size_t)EQ * TQ * 4)    // bf16 h: E*T*F*2 ~ 25 MB

#if __has_builtin(__builtin_amdgcn_rcpf)
#define RCPF(x) __builtin_amdgcn_rcpf(x)
#else
#define RCPF(x) (1.0f / (x))
#endif

static __device__ __forceinline__ unsigned short f2bf(float f) {
  unsigned u = __float_as_uint(f);
  u += 0x7FFFu + ((u >> 16) & 1u);   // round-to-nearest-even
  return (unsigned short)(u >> 16);
}

// packed pair fp32 -> bf16 (lo -> [15:0], hi -> [31:16]); v_cvt_pk_bf16_f32
static __device__ __forceinline__ unsigned pk_bf16(float lo, float hi) {
  float2 f2; f2.x = lo; f2.y = hi;
  __hip_bfloat162 h2 = __float22bfloat162_rn(f2);
  unsigned r;
  __builtin_memcpy(&r, &h2, sizeof(r));
  return r;
}

static __device__ __forceinline__ v16bf frag16(const unsigned short* p0,
                                               const unsigned short* p1) {
  v8bf lo = *reinterpret_cast<const v8bf*>(p0);
  v8bf hi = *reinterpret_cast<const v8bf*>(p1);
  return __builtin_shufflevector(lo, hi, 0,1,2,3,4,5,6,7,8,9,10,11,12,13,14,15);
}

// ---------------- fp32 -> bf16 conversion of hidden_states -----------------
__global__ __launch_bounds__(256) void cvt_x_bf16(const float* __restrict__ x,
                                                  unsigned int* __restrict__ xb) {
  size_t i = ((size_t)blockIdx.x * blockDim.x + threadIdx.x) * 8;
  float4 a = *reinterpret_cast<const float4*>(x + i);
  float4 b = *reinterpret_cast<const float4*>(x + i + 4);
  uint4 r;
  r.x = pk_bf16(a.x, a.y);
  r.y = pk_bf16(a.z, a.w);
  r.z = pk_bf16(b.x, b.y);
  r.w = pk_bf16(b.z, b.w);
  *reinterpret_cast<uint4*>(xb + i / 2) = r;
}

// ---------------- router: logits -> softmax -> top4 -> expert lists --------
__global__ __launch_bounds__(256) void router_topk(const float* __restrict__ x,
                                                   const float* __restrict__ gw,
                                                   int* __restrict__ counts,
                                                   int* __restrict__ list,
                                                   float* __restrict__ rweight) {
  const int warp = threadIdx.x >> 5, lane = threadIdx.x & 31;
  const int t = blockIdx.x * 8 + warp;

  float acc[EQ];
#pragma unroll
  for (int e = 0; e < EQ; ++e) acc[e] = 0.f;

  const float* xr = x + (size_t)t * DQ;
  for (int d = lane; d < DQ; d += 32) {
    float xv = xr[d];
    const float* g = gw + (size_t)d * EQ;
#pragma unroll
    for (int e = 0; e < EQ; ++e) acc[e] = fmaf(xv, g[e], acc[e]);
  }
#pragma unroll
  for (int e = 0; e < EQ; ++e) {
    float v = acc[e];
#pragma unroll
    for (int off = 16; off > 0; off >>= 1) v += __shfl_xor(v, off, 32);
    acc[e] = v;
  }
  float mx = acc[0];
#pragma unroll
  for (int e = 1; e < EQ; ++e) mx = fmaxf(mx, acc[e]);
  float p[EQ];
#pragma unroll
  for (int e = 0; e < EQ; ++e) p[e] = __expf(acc[e] - mx);

  int   idx[KQ];
  float val[KQ];
  float s = 0.f;
#pragma unroll
  for (int k = 0; k < KQ; ++k) {
    int best = 0; float bv = -1.f;
#pragma unroll
    for (int e = 0; e < EQ; ++e)
      if (p[e] > bv) { bv = p[e]; best = e; }
    idx[k] = best; val[k] = bv; p[best] = -1.f; s += bv;
  }
  if (lane == 0) {
    float inv = RCPF(s);   // softmax Z cancels in renormalization
#pragma unroll
    for (int k = 0; k < KQ; ++k) {
      int e = idx[k];
      int pos = atomicAdd(counts + e, 1);
      list[e * TQ + pos]    = t;
      rweight[e * TQ + pos] = val[k] * inv;
    }
  }
}

// ---------------- GEMM1: h = silu(x @ Wg) * (x @ Wu), per expert -----------
// Block tile M=128 x N=64 (of F), K-stage=64; 8 waves as 4(M)x2(N), each wave
// 32x32 for gate and up. Pipeline: A via global_load_async_to_lds_b128
// (ASYNCcnt), B via register prefetch + v_cvt_pk_bf16_f32 drain after WMMAs.
__global__ __launch_bounds__(256) void moe_gemm1(
    const unsigned short* __restrict__ xb, const float* __restrict__ wgu,
    const int* __restrict__ counts, const int* __restrict__ list,
    unsigned short* __restrict__ hbuf) {
  __shared__ unsigned short As[2][128 * LDAL];
  __shared__ unsigned short Bg[2][64 * LDAL];
  __shared__ unsigned short Bu[2][64 * LDAL];
  __shared__ int rows_s[128];

  const int e   = blockIdx.z;
  const int cnt = counts[e];
  const int m0  = blockIdx.y * 128;
  if (m0 >= cnt) return;
  const int n0  = blockIdx.x * 64;
  const int tid = threadIdx.x, warp = tid >> 5, lane = tid & 31;
  const int wm0 = (warp >> 1) * 32, wn0 = (warp & 1) * 32;

  if (tid < 128) {
    int m = m0 + tid;
    rows_s[tid] = (m < cnt) ? list[e * TQ + m] : list[e * TQ];
  }
  __syncthreads();

  const float* wB = wgu + (size_t)e * DQ * (2 * FQ);
  const int ar  = tid >> 3;        // A row base (slot i row = ar + i*32)
  const int ac8 = (tid & 7) * 8;   // A col group within K=64

  float2 pg[8], pu[8];             // B prefetch registers (64k x 64n tile)

  auto loadB = [&](int k0) {
#pragma unroll
    for (int i = 0; i < 8; ++i) {
      int lin = tid + i * 256;           // 0..2047 pairs
      int kk = lin >> 5, nn = (lin & 31) * 2;
      const float* row = wB + (size_t)(k0 + kk) * (2 * FQ) + n0 + nn;
      pg[i] = *reinterpret_cast<const float2*>(row);
      pu[i] = *reinterpret_cast<const float2*>(row + FQ);
    }
  };
  auto storeB = [&](int p) {
#pragma unroll
    for (int i = 0; i < 8; ++i) {
      int lin = tid + i * 256;
      int kk = lin >> 5, nn = (lin & 31) * 2;
      unsigned g2 = pk_bf16(pg[i].x, pg[i].y);
      unsigned u2 = pk_bf16(pu[i].x, pu[i].y);
      Bg[p][nn * LDAL + kk]       = (unsigned short)g2;
      Bg[p][(nn + 1) * LDAL + kk] = (unsigned short)(g2 >> 16);
      Bu[p][nn * LDAL + kk]       = (unsigned short)u2;
      Bu[p][(nn + 1) * LDAL + kk] = (unsigned short)(u2 >> 16);
    }
  };
  auto issueA = [&](int k0, int p) {
#pragma unroll
    for (int i = 0; i < 4; ++i) {
      int r = ar + i * 32;
      const unsigned short* src = xb + (size_t)rows_s[r] * DQ + k0 + ac8;
      unsigned dst = (unsigned)(uintptr_t)&As[p][r * LDAL + ac8];
      asm volatile("global_load_async_to_lds_b128 %0, %1, off"
                   :: "v"(dst), "v"(src) : "memory");
    }
  };

  v8f zero = {0.f,0.f,0.f,0.f,0.f,0.f,0.f,0.f};
  v8f accG[2][2], accU[2][2];
#pragma unroll
  for (int mi = 0; mi < 2; ++mi)
#pragma unroll
    for (int ni = 0; ni < 2; ++ni) { accG[mi][ni] = zero; accU[mi][ni] = zero; }

  // prologue: stage 0
  loadB(0);
  issueA(0, 0);
  storeB(0);
  asm volatile("s_wait_asynccnt 0x0" ::: "memory");
  __syncthreads();

  for (int k0 = 0, p = 0; k0 < DQ; k0 += 64, p ^= 1) {
    const bool more = (k0 + 64) < DQ;
    if (more) { loadB(k0 + 64); issueA(k0 + 64, p ^ 1); }

    const int mlo = (lane & 16) ? 16 : 0;
    const int klo = (lane & 16) ? 16 : 0;
#pragma unroll
    for (int ks = 0; ks < 64; ks += 32) {
      v16bf a[2], bg[2], bu[2];
#pragma unroll
      for (int mi = 0; mi < 2; ++mi) {
        const unsigned short* base =
            &As[p][(wm0 + mi * 16 + (lane & 15)) * LDAL + ks];
        a[mi] = frag16(base + mlo, base + 8 + mlo);
      }
#pragma unroll
      for (int ni = 0; ni < 2; ++ni) {
        const unsigned short* gb =
            &Bg[p][(wn0 + ni * 16 + (lane & 15)) * LDAL + ks] + klo;
        bg[ni] = frag16(gb, gb + 8);
        const unsigned short* ub =
            &Bu[p][(wn0 + ni * 16 + (lane & 15)) * LDAL + ks] + klo;
        bu[ni] = frag16(ub, ub + 8);
      }
#pragma unroll
      for (int mi = 0; mi < 2; ++mi)
#pragma unroll
        for (int ni = 0; ni < 2; ++ni) {
          accG[mi][ni] = __builtin_amdgcn_wmma_f32_16x16x32_bf16(
              false, a[mi], false, bg[ni], (short)0, accG[mi][ni], false, false);
          accU[mi][ni] = __builtin_amdgcn_wmma_f32_16x16x32_bf16(
              false, a[mi], false, bu[ni], (short)0, accU[mi][ni], false, false);
        }
    }

    if (more) storeB(p ^ 1);
    asm volatile("s_wait_asynccnt 0x0" ::: "memory");
    __syncthreads();
  }

  // epilogue: h = silu(g) * u = g * u * rcp(1 + exp(-g)), bf16 store
  const int rofs = (lane & 16) ? 8 : 0;
#pragma unroll
  for (int mi = 0; mi < 2; ++mi)
#pragma unroll
    for (int ni = 0; ni < 2; ++ni) {
      int col = n0 + wn0 + ni * 16 + (lane & 15);
#pragma unroll
      for (int r = 0; r < 8; ++r) {
        int rl = wm0 + mi * 16 + r + rofs;
        float g = accG[mi][ni][r];
        float u = accU[mi][ni][r];
        float h = g * u * RCPF(1.f + __expf(-g));
        hbuf[((size_t)e * TQ + m0 + rl) * FQ + col] = f2bf(h);
      }
    }
}

// ---------------- GEMM2: out[t] += w_t * (h @ Wd), per expert --------------
__global__ __launch_bounds__(256) void moe_gemm2(
    const unsigned short* __restrict__ hbuf, const float* __restrict__ wd,
    const int* __restrict__ counts, const int* __restrict__ list,
    const float* __restrict__ rweight, float* __restrict__ out) {
  __shared__ unsigned short As[2][128 * LDAL];
  __shared__ unsigned short Bs[2][64 * LDAL];
  __shared__ int   tok_s[128];
  __shared__ float w_s[128];

  const int e   = blockIdx.z;
  const int cnt = counts[e];
  const int m0  = blockIdx.y * 128;
  if (m0 >= cnt) return;
  const int n0  = blockIdx.x * 64;
  const int tid = threadIdx.x, warp = tid >> 5, lane = tid & 31;
  const int wm0 = (warp >> 1) * 32, wn0 = (warp & 1) * 32;

  if (tid < 128) {
    int m = m0 + tid;
    tok_s[tid] = (m < cnt) ? list[e * TQ + m] : 0;
    w_s[tid]   = (m < cnt) ? rweight[e * TQ + m] : 0.f;
  }
  __syncthreads();

  const float* wB = wd + (size_t)e * FQ * DQ;
  const unsigned short* hA = hbuf + (size_t)e * TQ * FQ;
  const int ar  = tid >> 3;
  const int ac8 = (tid & 7) * 8;

  float2 pb[8];

  auto loadB = [&](int k0) {
#pragma unroll
    for (int i = 0; i < 8; ++i) {
      int lin = tid + i * 256;
      int kk = lin >> 5, nn = (lin & 31) * 2;
      pb[i] = *reinterpret_cast<const float2*>(wB + (size_t)(k0 + kk) * DQ + n0 + nn);
    }
  };
  auto storeB = [&](int p) {
#pragma unroll
    for (int i = 0; i < 8; ++i) {
      int lin = tid + i * 256;
      int kk = lin >> 5, nn = (lin & 31) * 2;
      unsigned b2 = pk_bf16(pb[i].x, pb[i].y);
      Bs[p][nn * LDAL + kk]       = (unsigned short)b2;
      Bs[p][(nn + 1) * LDAL + kk] = (unsigned short)(b2 >> 16);
    }
  };
  auto issueA = [&](int k0, int p) {
#pragma unroll
    for (int i = 0; i < 4; ++i) {
      int r = ar + i * 32;
      const unsigned short* src = hA + (size_t)(m0 + r) * FQ + k0 + ac8;
      unsigned dst = (unsigned)(uintptr_t)&As[p][r * LDAL + ac8];
      asm volatile("global_load_async_to_lds_b128 %0, %1, off"
                   :: "v"(dst), "v"(src) : "memory");
    }
  };

  v8f zero = {0.f,0.f,0.f,0.f,0.f,0.f,0.f,0.f};
  v8f acc[2][2];
#pragma unroll
  for (int mi = 0; mi < 2; ++mi)
#pragma unroll
    for (int ni = 0; ni < 2; ++ni) acc[mi][ni] = zero;

  loadB(0);
  issueA(0, 0);
  storeB(0);
  asm volatile("s_wait_asynccnt 0x0" ::: "memory");
  __syncthreads();

  for (int k0 = 0, p = 0; k0 < FQ; k0 += 64, p ^= 1) {
    const bool more = (k0 + 64) < FQ;
    if (more) { loadB(k0 + 64); issueA(k0 + 64, p ^ 1); }

    const int mlo = (lane & 16) ? 16 : 0;
    const int klo = (lane & 16) ? 16 : 0;
#pragma unroll
    for (int ks = 0; ks < 64; ks += 32) {
      v16bf a[2], b[2];
#pragma unroll
      for (int mi = 0; mi < 2; ++mi) {
        const unsigned short* base =
            &As[p][(wm0 + mi * 16 + (lane & 15)) * LDAL + ks];
        a[mi] = frag16(base + mlo, base + 8 + mlo);
      }
#pragma unroll
      for (int ni = 0; ni < 2; ++ni) {
        const unsigned short* bb =
            &Bs[p][(wn0 + ni * 16 + (lane & 15)) * LDAL + ks] + klo;
        b[ni] = frag16(bb, bb + 8);
      }
#pragma unroll
      for (int mi = 0; mi < 2; ++mi)
#pragma unroll
        for (int ni = 0; ni < 2; ++ni)
          acc[mi][ni] = __builtin_amdgcn_wmma_f32_16x16x32_bf16(
              false, a[mi], false, b[ni], (short)0, acc[mi][ni], false, false);
    }

    if (more) storeB(p ^ 1);
    asm volatile("s_wait_asynccnt 0x0" ::: "memory");
    __syncthreads();
  }

  const int rofs = (lane & 16) ? 8 : 0;
#pragma unroll
  for (int mi = 0; mi < 2; ++mi)
#pragma unroll
    for (int ni = 0; ni < 2; ++ni) {
      int col = n0 + wn0 + ni * 16 + (lane & 15);
#pragma unroll
      for (int r = 0; r < 8; ++r) {
        int rl = wm0 + mi * 16 + r + rofs;
        if (m0 + rl < cnt) {
          float v = acc[mi][ni][r] * w_s[rl];
          unsafeAtomicAdd(out + (size_t)tok_s[rl] * DQ + col, v);
        }
      }
    }
}

extern "C" void kernel_launch(void* const* d_in, const int* in_sizes, int n_in,
                              void* d_out, int out_size, void* d_ws, size_t ws_size,
                              hipStream_t stream) {
  const float* x   = (const float*)d_in[0];   // (T,D)
  const float* gw  = (const float*)d_in[1];   // (D,E)
  const float* wgu = (const float*)d_in[2];   // (E,D,2F)
  const float* wdn = (const float*)d_in[3];   // (E,F,D)
  float* out = (float*)d_out;                 // (T,D)

  char* ws = (char*)d_ws;
  unsigned int*   xbf32 = (unsigned int*)(ws + X_OFF);
  unsigned short* xbf   = (unsigned short*)(ws + X_OFF);
  int*            cnts  = (int*)(ws + CNT_OFF);
  int*            list  = (int*)(ws + LIST_OFF);
  float*          rwt   = (float*)(ws + RW_OFF);
  unsigned short* hbuf  = (unsigned short*)(ws + H_OFF);

  (void)hipMemsetAsync(cnts, 0, 256, stream);
  (void)hipMemsetAsync(out, 0, (size_t)TQ * DQ * sizeof(float), stream);

  cvt_x_bf16<<<(TQ * DQ) / (256 * 8), 256, 0, stream>>>(x, xbf32);
  router_topk<<<TQ / 8, 256, 0, stream>>>(x, gw, cnts, list, rwt);
  moe_gemm1<<<dim3(FQ / 64, TQ / 128, EQ), 256, 0, stream>>>(xbf, wgu, cnts, list, hbuf);
  moe_gemm2<<<dim3(DQ / 64, TQ / 128, EQ), 256, 0, stream>>>(hbuf, wdn, cnts, list, rwt, out);
}
